// SpikeLayer_29240137351456
// MI455X (gfx1250) — matrix operand order repaired and checked by reading, not compiled, not run on hardware
//
#include <hip/hip_runtime.h>

// ---- CDNA5 vector types ----
typedef __attribute__((ext_vector_type(16))) __bf16 v16bf;
typedef __attribute__((ext_vector_type(8)))  __bf16 v8bf;
typedef __attribute__((ext_vector_type(8)))  float  v8f;

// Problem constants (fixed by the reference)
#define NB    32
#define CIN   128
#define HH    56
#define WW    56
#define COUT  256
#define NHW   (NB * HH * WW)      // 100352
#define KTOT  (CIN * 9)           // 1152
#define XCH   (HH * WW * 2)       // 6272: x stride per input channel (elements)
#define XNS   (CIN * XCH)         // 802816: x stride per image (elements)

// Tiling
#define BM 128
#define BN 64
#define BK 32
#define THREADS 256
#define NIT (9 * (CIN / BK))      // 36 K-chunks: tap-major, ci-minor

// Spike constants: o = ((u - Vth) > Vth)  ->  u > 2*Vth
#define U_THRESH 0.4f
#define TAU      0.2f

union AFrag { v16bf v; v8bf h[2]; };

__global__ __launch_bounds__(THREADS)
void snn_conv2_wmma_bf16(const float* __restrict__ x,   // [32,128,56,56,2]
                         const float* __restrict__ Wt,  // [256,128,3,3]
                         const float* __restrict__ bias,// [256]
                         float* __restrict__ out)       // [32,256,56,56,2]
{
    // Double-buffered LDS: 16 KB A + 16 KB B = 32 KB of the 320 KB WGP pool
    __shared__ __align__(16) __bf16 As[2][BM][BK];
    __shared__ __align__(16) __bf16 Bs[2][2][BN][BK];

    const int tid  = threadIdx.x;
    const int lane = tid & 31;
    const int wave = tid >> 5;      // 0..7
    const int wm   = wave >> 2;     // 0..1 -> 64 M-rows each
    const int wn   = wave & 3;      // 0..3 -> 16 N-cols each

    const int mb       = blockIdx.x & 1;       // 2 M-blocks
    const int nb       = blockIdx.x >> 1;      // 1568 N-blocks
    const int m_base   = mb * BM;
    const int col_base = nb * BN;

    // ---------- per-thread staging roles (fixed for whole kernel) ----------
    // B: this thread owns (col c_, step s_) and 16 consecutive ci values
    const int c_     = tid & (BN - 1);
    const int s_     = (tid >> 6) & 1;
    const int kciB   = (tid >> 7) * 16;        // local k offset in tile
    const int colg   = col_base + c_;
    const int n_     = colg / (HH * WW);
    const int r_     = colg - n_ * (HH * WW);
    const int h_     = r_ / WW;
    const int w_     = r_ - h_ * WW;
    const int xbase0 = n_ * XNS + s_ + kciB * XCH;  // ci fold: + (cc+i)*XCH later
    // A: this thread owns weight row rowA and 16 consecutive ci values
    const int rowA   = tid & (BM - 1);
    const int kciA   = (tid >> 7) * 16;
    const int wbase0 = (m_base + rowA) * KTOT + kciA * 9;  // + cc*9 + i*9 + q later

    // issue global loads for K-chunk `nit` (tap q = nit/4, ci base cc = (nit%4)*32)
    auto issue_loads = [&](int nit, float* a_r, float* b_r) {
        if (nit >= NIT) return;
        const int q  = nit >> 2;
        const int cc = (nit & 3) * BK;
        const int kh = q / 3;
        const int kw = q - kh * 3;
        const int ih = h_ + kh - 1;
        const int iw = w_ + kw - 1;
        const bool valid = ((unsigned)ih < (unsigned)HH) && ((unsigned)iw < (unsigned)WW);
        const int boff = xbase0 + ih * (WW * 2) + iw * 2 + cc * XCH;
        const int woff = wbase0 + q + cc * 9;
#pragma unroll
        for (int i = 0; i < 16; ++i)            // clause of global_load_b32, imm offsets
            b_r[i] = valid ? x[boff + i * XCH] : 0.0f;
#pragma unroll
        for (int i = 0; i < 16; ++i)
            a_r[i] = Wt[woff + i * 9];
    };

    // convert + store K-chunk `nit` into LDS buffer `buf`
    auto store_tiles = [&](int nit, int buf, const float* a_r, const float* b_r) {
        if (nit >= NIT) return;
        v8bf pb0, pb1, pa0, pa1;
#pragma unroll
        for (int i = 0; i < 8; ++i) {
            pb0[i] = (__bf16)b_r[i];
            pb1[i] = (__bf16)b_r[8 + i];
            pa0[i] = (__bf16)a_r[i];
            pa1[i] = (__bf16)a_r[8 + i];
        }
        v8bf* bq = (v8bf*)&Bs[buf][s_][c_][kciB];
        bq[0] = pb0; bq[1] = pb1;                // 2x ds_store_b128
        v8bf* aq = (v8bf*)&As[buf][rowA][kciA];
        aq[0] = pa0; aq[1] = pa1;
    };

    // ---------- accumulators ----------
    const v8f vzero = {0.f, 0.f, 0.f, 0.f, 0.f, 0.f, 0.f, 0.f};
    v8f acc0[4], acc1[4];
#pragma unroll
    for (int i = 0; i < 4; ++i) { acc0[i] = vzero; acc1[i] = vzero; }

    // per-lane fragment addressing (contiguous k order, same for A and B: half h holds k = 16h..16h+15)
    const int bcol = wn * 16 + (lane & 15);
    const int bko  = (lane >> 4) * 16;  // B: lanes 0-15 K 0..15, lanes 16-31 K 16..31
    const int ako  = (lane >> 4) * 16;  // A: lanes 0-15 K 0..15, lanes 16-31 K 16..31
    const int arow = wm * 64 + (lane & 15);

    // ---------- software pipeline: prologue stages chunk 0 into buf 0 ----------
    {
        float a_r[16], b_r[16];
        issue_loads(0, a_r, b_r);
        store_tiles(0, 0, a_r, b_r);
    }
    __syncthreads();

    for (int it = 0; it < NIT; ++it) {
        const int cur = it & 1;

        // start next chunk's global loads (overlap with WMMA below)
        float a_n[16], b_n[16];
        issue_loads(it + 1, a_n, b_n);

        // ---- compute chunk `it` from buf[cur] ----
        AFrag bf0, bf1;
        const v8bf* bp0 = (const v8bf*)&Bs[cur][0][bcol][bko];
        const v8bf* bp1 = (const v8bf*)&Bs[cur][1][bcol][bko];
        bf0.h[0] = bp0[0]; bf0.h[1] = bp0[1];    // 2x ds_load_b128
        bf1.h[0] = bp1[0]; bf1.h[1] = bp1[1];
#pragma unroll
        for (int mt = 0; mt < 4; ++mt) {
            const v8bf* ap = (const v8bf*)&As[cur][arow + mt * 16][ako];
            AFrag af;
            af.h[0] = ap[0];                     // K = ako..ako+7
            af.h[1] = ap[1];                     // K = ako+8..ako+15
            acc0[mt] = __builtin_amdgcn_wmma_f32_16x16x32_bf16(
                false, af.v, false, bf0.v, (short)0, acc0[mt], false, false);
            acc1[mt] = __builtin_amdgcn_wmma_f32_16x16x32_bf16(
                false, af.v, false, bf1.v, (short)0, acc1[mt], false, false);
        }

        // ---- stage next chunk into the other buffer, then one barrier ----
        store_tiles(it + 1, cur ^ 1, a_n, b_n);
        __syncthreads();
    }

    // ---------- epilogue: bias + 2-step membrane recurrence + spike ----------
    {
        const int col = col_base + wn * 16 + (lane & 15);   // D: N = lane%16
        const int n   = col / (HH * WW);
        const int r   = col - n * (HH * WW);
        const int h   = r / WW;
        const int w   = r - h * WW;
        const int mhi = (lane >> 4) * 8;                    // D: VGPR j -> M = j (+8 hi lanes)
#pragma unroll
        for (int mt = 0; mt < 4; ++mt) {
#pragma unroll
            for (int j = 0; j < 8; ++j) {
                int co   = m_base + wm * 64 + mt * 16 + j + mhi;
                float bb = bias[co];
                float u1 = acc0[mt][j] + bb;
                float o1 = (u1 > U_THRESH) ? 1.0f : 0.0f;
                float u2 = TAU * u1 * (1.0f - o1) + acc1[mt][j] + bb;
                float o2 = (u2 > U_THRESH) ? 1.0f : 0.0f;
                size_t oi = ((((size_t)n * COUT + co) * HH + h) * WW + w) * 2;
                float2 ov = make_float2(o1, o2);
                *(float2*)(&out[oi]) = ov;                  // global_store_b64, 8B aligned
            }
        }
    }
}

extern "C" void kernel_launch(void* const* d_in, const int* in_sizes, int n_in,
                              void* d_out, int out_size, void* d_ws, size_t ws_size,
                              hipStream_t stream) {
    (void)in_sizes; (void)n_in; (void)d_ws; (void)ws_size; (void)out_size;
    const float* x  = (const float*)d_in[0];
    const float* Wt = (const float*)d_in[1];
    const float* b  = (const float*)d_in[2];
    float* out = (float*)d_out;

    const int grid = (NHW / BN) * (COUT / BM);   // 1568 * 2 = 3136 blocks
    snn_conv2_wmma_bf16<<<grid, THREADS, 0, stream>>>(x, Wt, b, out);
}
